// TestModel_42752104464710
// MI455X (gfx1250) — 8-bit matrix operand order repaired and checked by reading, compiled, not run on hardware
//
#include <hip/hip_runtime.h>

typedef __attribute__((ext_vector_type(16))) int   v16i;
typedef __attribute__((ext_vector_type(8)))  int   v8i;
typedef __attribute__((ext_vector_type(8)))  float v8f;

#define ABDIM 2048
#define CDIM  512
#define DDIM  32
#define EDIM  128
#define NDIM  4096

__device__ __forceinline__ unsigned char f32_to_e4m3(float x) {
    unsigned int u = __float_as_uint(x);
    unsigned char sign = (unsigned char)((u >> 24) & 0x80u);
    float ax = __uint_as_float(u & 0x7FFFFFFFu);
    if (ax != ax) return (unsigned char)(sign | 0x7Fu);
    if (ax >= 448.0f) return (unsigned char)(sign | 0x7Eu);
    if (ax < 0.015625f) {
        int m = (int)(ax * 512.0f + 0.5f);
        return (unsigned char)(sign | (unsigned char)m);
    }
    unsigned int au = __float_as_uint(ax);
    int E = (int)((au >> 23) & 0xFFu) - 127;
    unsigned int m23  = au & 0x7FFFFFu;
    unsigned int mant = m23 >> 20;
    unsigned int rest = m23 & 0xFFFFFu;
    if (rest > 0x80000u || (rest == 0x80000u && (mant & 1u))) ++mant;
    if (mant == 8u) { mant = 0u; ++E; if (E > 8) return (unsigned char)(sign | 0x7Eu); }
    return (unsigned char)(sign | ((unsigned int)(E + 7) << 3) | mant);
}

__device__ __forceinline__ float slot_scale(const float* amax, int i) {
    return fmaxf(amax[i] * (1.0f / 448.0f), 1e-12f);
}

__global__ void init_kernel(float* amax) {
    if (threadIdx.x < 8) amax[threadIdx.x] = 0.0f;
}

__global__ void amax_kernel(const float* __restrict__ src, int n, float* slot) {
    float m = 0.0f;
    for (int i = blockIdx.x * blockDim.x + threadIdx.x; i < n; i += gridDim.x * blockDim.x)
        m = fmaxf(m, fabsf(src[i]));
    #pragma unroll
    for (int off = 16; off > 0; off >>= 1)
        m = fmaxf(m, __shfl_xor(m, off, 32));
    if ((threadIdx.x & 31) == 0)
        atomicMax((unsigned int*)slot, __float_as_uint(m));
}

__global__ void quant_kernel(const float* __restrict__ src, unsigned char* __restrict__ dst,
                             int n, const float* __restrict__ amax, int slot) {
    float inv = 1.0f / slot_scale(amax, slot);
    int i = blockIdx.x * blockDim.x + threadIdx.x;
    if (i < n) dst[i] = f32_to_e4m3(src[i] * inv);
}

__global__ void quant_w2t_kernel(const float* __restrict__ w2, unsigned char* __restrict__ dst,
                                 const float* __restrict__ amax) {
    float inv = 1.0f / slot_scale(amax, 2);
    int j = blockIdx.x * blockDim.x + threadIdx.x;
    if (j < EDIM * DDIM * CDIM) {
        int c = j & 511, d = (j >> 9) & 31, e = j >> 14;
        dst[j] = f32_to_e4m3(w2[(c << 12) + (d << 7) + e] * inv);
    }
}

__global__ void quant_ht_kernel(const float* __restrict__ h, unsigned char* __restrict__ dst,
                                const float* __restrict__ amax) {
    float inv = 1.0f / slot_scale(amax, 3);
    int j = blockIdx.x * blockDim.x + threadIdx.x;
    if (j < EDIM * ABDIM * DDIM) {
        int d = j & 31, ab = (j >> 5) & 2047, e = j >> 16;
        dst[j] = f32_to_e4m3(h[(ab << 12) + (d << 7) + e] * inv);
    }
}

__global__ void __launch_bounds__(256) gemm1_kernel(
        const unsigned char* __restrict__ Aq, const unsigned char* __restrict__ Bq,
        float* __restrict__ H, const float* __restrict__ amax, float* __restrict__ amax_h) {
    int wave = threadIdx.x >> 5;
    int lane = threadIdx.x & 31;
    int tile = blockIdx.x * 8 + wave;
    int m0 = (tile >> 8) << 4;
    int n0 = (tile & 255) << 4;
    float sc = slot_scale(amax, 0) * slot_scale(amax, 1);

    int M = m0 + (lane & 15);
    int kbase = (lane < 16) ? 0 : 8;
    v8f acc = {};
    for (int kk = 0; kk < CDIM; kk += 128) {
        v16i a;
        const unsigned char* ap = Aq + M * CDIM + kk + kbase;
        #pragma unroll
        for (int h2 = 0; h2 < 2; ++h2)
            #pragma unroll
            for (int j = 0; j < 4; ++j) {
                int2 p = *(const int2*)(ap + h2 * 64 + j * 16);
                a[h2 * 8 + j * 2] = p.x; a[h2 * 8 + j * 2 + 1] = p.y;
            }
        v16i b;
        #pragma unroll
        for (int g = 0; g < 16; ++g) {
            unsigned int w = 0;
            for (int t = 0; t < 4; ++t) {
                int s = 4 * g + t;
                int k = kk + kbase + ((s >> 5) << 6) + (((s >> 3) & 3) << 4) + (s & 7);
                w |= (unsigned int)Bq[(size_t)k * NDIM + n0 + (lane & 15)] << (8 * t);
            }
            b[g] = (int)w;
        }
        acc = __builtin_amdgcn_wmma_f32_16x16x128_fp8_fp8(a, b, (short)0, acc, false, false);
    }
    float lmax = 0.0f;
    int row0 = m0 + ((lane < 16) ? 0 : 8);
    int col  = n0 + (lane & 15);
    #pragma unroll
    for (int r = 0; r < 8; ++r) {
        float v = acc[r] * sc;
        lmax = fmaxf(lmax, fabsf(v));
        H[(size_t)(row0 + r) * NDIM + col] = v;
    }
    #pragma unroll
    for (int off = 16; off > 0; off >>= 1)
        lmax = fmaxf(lmax, __shfl_xor(lmax, off, 32));
    if (lane == 0) atomicMax((unsigned int*)amax_h, __float_as_uint(lmax));
}

__global__ void __launch_bounds__(256) gemm2_kernel(
        const unsigned char* __restrict__ Hq,
        const unsigned char* __restrict__ W2q,
        float* __restrict__ Out,
        const float* __restrict__ amax) {
    __shared__ float lds[16 * 16 * 8];
    int wave = threadIdx.x >> 5;
    int lane = threadIdx.x & 31;
    int m0 = (blockIdx.x >> 5) << 4;
    int n0 = (blockIdx.x & 31) << 4;
    float sc = slot_scale(amax, 3) * slot_scale(amax, 2);

    int M = m0 + (lane & 15);
    int kbase = (lane < 16) ? 0 : 8;
    int row0  = (lane < 16) ? 0 : 8;
    int col   = lane & 15;

    for (int ee = 0; ee < EDIM; ee += 8) {
        int e = ee + wave;
        v8i a = {};
        const unsigned char* ap = Hq + (((size_t)e * ABDIM + M) << 5) + kbase;
        int2 p0 = *(const int2*)(ap);
        int2 p1 = *(const int2*)(ap + 16);
        a[0] = p0.x; a[1] = p0.y; a[2] = p1.x; a[3] = p1.y;
        v8i b = {};
        for (int g = 0; g < 4; ++g) {
            unsigned int w = 0;
            for (int t = 0; t < 4; ++t) {
                int s = 4 * g + t;
                int d = kbase + ((s >> 3) << 4) + (s & 7);
                w |= (unsigned int)W2q[(((size_t)e * DDIM + d) << 9) + n0 + (lane & 15)] << (8 * t);
            }
            b[g] = (int)w;
        }
        v8f c = {};
        c = __builtin_amdgcn_wmma_f32_16x16x64_fp8_fp8(a, b, (short)0, c, false, false);
        #pragma unroll
        for (int r = 0; r < 8; ++r)
            lds[((row0 + r) * 16 + col) * 8 + wave] = c[r] * sc;
        __syncthreads();
        {
            int pr = threadIdx.x >> 4;
            int pc = threadIdx.x & 15;
            float4 v0 = *(float4*)&lds[(pr * 16 + pc) * 8 + 0];
            float4 v1 = *(float4*)&lds[(pr * 16 + pc) * 8 + 4];
            float* op = Out + (((size_t)(m0 + pr) * CDIM) + (n0 + pc)) * EDIM + ee;
            *(float4*)op = v0;
            *(float4*)(op + 4) = v1;
        }
        __syncthreads();
    }
}

extern "C" void kernel_launch(void* const* d_in, const int* in_sizes, int n_in,
                              void* d_out, int out_size, void* d_ws, size_t ws_size,
                              hipStream_t stream) {
    const float* x  = (const float*)d_in[0];
    const float* w1 = (const float*)d_in[1];
    const float* w2 = (const float*)d_in[2];
    float* out = (float*)d_out;

    char* ws = (char*)d_ws;
    float*         amax = (float*)ws;
    unsigned char* xq   = (unsigned char*)(ws + 256);
    unsigned char* w1q  = xq  + (size_t)ABDIM * CDIM;
    unsigned char* w2qt = w1q + (size_t)CDIM * NDIM;
    unsigned char* hqt  = w2qt + (size_t)EDIM * DDIM * CDIM;
    float*         h    = (float*)(hqt + (size_t)EDIM * ABDIM * DDIM);

    const int nx  = ABDIM * CDIM;
    const int nw  = CDIM * DDIM * EDIM;
    const int nh  = ABDIM * NDIM;

    init_kernel<<<1, 32, 0, stream>>>(amax);
    amax_kernel<<<256, 256, 0, stream>>>(x,  nx, amax + 0);
    amax_kernel<<<256, 256, 0, stream>>>(w1, nw, amax + 1);
    amax_kernel<<<256, 256, 0, stream>>>(w2, nw, amax + 2);

    quant_kernel<<<(nx + 255) / 256, 256, 0, stream>>>(x,  xq,  nx, amax, 0);
    quant_kernel<<<(nw + 255) / 256, 256, 0, stream>>>(w1, w1q, nw, amax, 1);
    quant_w2t_kernel<<<(nw + 255) / 256, 256, 0, stream>>>(w2, w2qt, amax);

    gemm1_kernel<<<4096, 256, 0, stream>>>(xq, w1q, h, amax, amax + 3);
    quant_ht_kernel<<<(nh + 255) / 256, 256, 0, stream>>>(h, hqt, amax);
    gemm2_kernel<<<4096, 256, 0, stream>>>(hqt, w2qt, out, amax);
}
